// SS2D_55559696941517
// MI455X (gfx1250) — matrix operand order repaired and checked by reading, not compiled, not run on hardware
//
#include <hip/hip_runtime.h>

// ---------------------------------------------------------------------------
// SS2D (VMamba) block for MI455X / gfx1250 — wave32, WMMA bf16 GEMMs
//   B=8, H=W=64, L=4096, D_MODEL=96, D_INNER=192, D_STATE=16, K=4, DT_RANK=6
// ---------------------------------------------------------------------------

#define BB      8
#define HH      64
#define WW      64
#define LL      4096          // H*W
#define DM      96            // d_model
#define DI      192           // d_inner
#define DS      16            // d_state
#define KD      4             // directions
#define DTR     6             // dt_rank
#define MROWS   224           // DI + 2*DS rows of fused projection matrix
#define MTOT    (BB * LL)     // 32768 pixel rows

typedef __attribute__((ext_vector_type(16))) __bf16        v16bf;
typedef __attribute__((ext_vector_type(8)))  float         v8f;
typedef __attribute__((ext_vector_type(4)))  unsigned int  u32x4;

__device__ __forceinline__ float dev_silu(float x) { return x / (1.0f + __expf(-x)); }

// Direction index transform: maps scan step l -> row of channel-last xc[B,L,C].
// k bit1 = flipped sequence, k bit0 = WH (transposed) order.
__device__ __forceinline__ int scan_row(int k, int l) {
    int t = (k & 2) ? (LL - 1 - l) : l;
    if (k & 1) t = ((t & 63) << 6) | (t >> 6);   // (w,h) <-> (h,w), H=W=64
    return t;
}

// One 16x16 f32 tile, A row-major [16,K] via pa, B as transposed [16,K] via pb.
// pa = Abase + (lane&15)*lda + 8*(lane>>4)   (bf16 elements)
// pb = Bbase + (lane&15)*ldb + 8*(lane>>4)
template <int KTOT>
__device__ __forceinline__ v8f wmma_bf16_tile(const __bf16* pa, const __bf16* pb) {
    v8f acc = {};
#pragma unroll
    for (int k0 = 0; k0 < KTOT; k0 += 32) {
        union { v16bf v; u32x4 u[2]; } a, b;
        a.u[0] = *(const u32x4*)(pa + k0);        // K = k0 + 8*half .. +7
        a.u[1] = *(const u32x4*)(pa + k0 + 16);   // K = k0 + 16 + 8*half .. +7
        b.u[0] = *(const u32x4*)(pb + k0);        // K = k0 + 8*half .. +7
        b.u[1] = *(const u32x4*)(pb + k0 + 16);   // K = k0 + 16 + 8*half .. +7
        acc = __builtin_amdgcn_wmma_f32_16x16x32_bf16(
            false, a.v, false, b.v, (short)0, acc, false, false);
    }
    return acc;
}

// ---------------------------------------------------------------------------
// Small prep kernels
// ---------------------------------------------------------------------------
__global__ void k_cvt_bf16(const float* __restrict__ src, __bf16* __restrict__ dst, int n) {
    int i = blockIdx.x * 256 + threadIdx.x;
    if (i < n) dst[i] = (__bf16)src[i];
}

// Build fused per-direction projection Wall[k] (224 x 192):
//   rows 0..191  : Weff = dt_projs_weight[k] (192x6) @ x_proj_weight[k][:6] (6x192)
//   rows 192..207: B-rows  (x_proj_weight rows 6..21)
//   rows 208..223: C-rows  (x_proj_weight rows 22..37)
__global__ void k_build_wall(const float* __restrict__ xpw, const float* __restrict__ dtw,
                             __bf16* __restrict__ wall) {
    int i = blockIdx.x * 256 + threadIdx.x;
    if (i >= KD * MROWS * DI) return;
    int c = i % DI;
    int m = (i / DI) % MROWS;
    int k = i / (MROWS * DI);
    float v;
    if (m < DI) {
        v = 0.0f;
#pragma unroll
        for (int r = 0; r < DTR; ++r)
            v += dtw[(k * DI + m) * DTR + r] * xpw[((size_t)k * 38 + r) * DI + c];
    } else {
        v = xpw[((size_t)k * 38 + DTR + (m - DI)) * DI + c];
    }
    wall[i] = (__bf16)v;
}

// ---------------------------------------------------------------------------
// GEMM 1: xz = x @ in_proj_w^T     M=32768, N=384, K=96
//   writes xq (cols 0..191) and z (cols 192..383) as f32 [M,192]
// ---------------------------------------------------------------------------
__global__ void k_inproj(const __bf16* __restrict__ xbf, const __bf16* __restrict__ wbf,
                         float* __restrict__ xq, float* __restrict__ z) {
    int lane = threadIdx.x & 31, wave = threadIdx.x >> 5;
    int mt = blockIdx.x;              // 0..2047
    int nt = blockIdx.y * 4 + wave;   // 0..23
    int r = lane & 15, half = lane >> 4;
    const __bf16* pa = xbf + ((size_t)(mt * 16 + r)) * DM + 8 * half;
    const __bf16* pb = wbf + ((size_t)(nt * 16 + r)) * DM + 8 * half;
    v8f acc = wmma_bf16_tile<DM>(pa, pb);
    int n = nt * 16 + r;
    int m0 = mt * 16 + half * 8;
    float* dst = (n < DI) ? xq : z;
    int nc = (n < DI) ? n : (n - DI);
#pragma unroll
    for (int i = 0; i < 8; ++i) dst[(size_t)(m0 + i) * DI + nc] = acc[i];
}

// ---------------------------------------------------------------------------
// Depthwise 3x3 SAME conv + bias + SiLU on channel-last xq[B,L,192]
//   writes xc_f32 and xc_bf16
// ---------------------------------------------------------------------------
__global__ void k_conv(const float* __restrict__ xq, const float* __restrict__ cw,
                       const float* __restrict__ cb, float* __restrict__ xcf,
                       __bf16* __restrict__ xcb) {
    size_t i = (size_t)blockIdx.x * 256 + threadIdx.x;
    if (i >= (size_t)BB * LL * DI) return;
    int c = (int)(i % DI);
    size_t bl = i / DI;
    int l = (int)(bl % LL);
    int b = (int)(bl / LL);
    int h = l >> 6, w = l & 63;
    float acc = cb[c];
#pragma unroll
    for (int kh = 0; kh < 3; ++kh) {
        int hh = h + kh - 1;
        if (hh < 0 || hh >= HH) continue;
#pragma unroll
        for (int kw = 0; kw < 3; ++kw) {
            int ww = w + kw - 1;
            if (ww < 0 || ww >= WW) continue;
            acc += cw[c * 9 + kh * 3 + kw] *
                   xq[((size_t)b * LL + (hh << 6) + ww) * DI + c];
        }
    }
    acc = dev_silu(acc);
    xcf[i] = acc;
    xcb[i] = (__bf16)acc;
}

// ---------------------------------------------------------------------------
// GEMM 2: per (b,k): Wall[k] (224x192) @ xs_k (192x4096)
//   B operand rows are permuted rows of xc_bf16 (contiguous K=192 each).
//   Output stored transposed: gout[b,k][l][224]  (delta | B16 | C16) f32
// ---------------------------------------------------------------------------
__global__ void k_gemm2(const __bf16* __restrict__ xcbf, const __bf16* __restrict__ wall,
                        float* __restrict__ gout) {
    int lane = threadIdx.x & 31, wave = threadIdx.x >> 5;
    int bk = blockIdx.x;              // b*4+k
    int b = bk >> 2, k = bk & 3;
    int mt = blockIdx.y;              // 0..13
    int nt = blockIdx.z * 4 + wave;   // 0..255
    int r = lane & 15, half = lane >> 4;
    int l = nt * 16 + r;
    const __bf16* pa = wall + ((size_t)k * MROWS + mt * 16 + r) * DI + 8 * half;
    const __bf16* pb = xcbf + ((size_t)b * LL + scan_row(k, l)) * DI + 8 * half;
    v8f acc = wmma_bf16_tile<DI>(pa, pb);
    float* p = gout + ((size_t)bk * LL + l) * MROWS + mt * 16 + half * 8;
#pragma unroll
    for (int i = 0; i < 8; ++i) p[i] = acc[i];
}

// ---------------------------------------------------------------------------
// Selective scan: one thread per (b,k,d); 16-state recurrence in registers.
//   h = exp(dt*A)*h + (dt*u)*B ;  y = h.C + u*Dp
//   Writes per-direction contribution back through the index involution.
// ---------------------------------------------------------------------------
__global__ void k_scan(const float* __restrict__ gout, const float* __restrict__ xcf,
                       const float* __restrict__ alog, const float* __restrict__ Dsv,
                       const float* __restrict__ dtb, float* __restrict__ ycon) {
    int bk = blockIdx.x;              // 0..31
    int b = bk >> 2, k = bk & 3;
    int d = threadIdx.x;              // 0..191
    float A[DS], h[DS];
#pragma unroll
    for (int n = 0; n < DS; ++n) {
        A[n] = -__expf(alog[((size_t)(k * DI + d)) * DS + n]);
        h[n] = 0.0f;
    }
    float bias = dtb[k * DI + d];
    float Dp = Dsv[k * DI + d];
    const float* grow = gout + (size_t)bk * LL * MROWS;
    const float* xb = xcf + (size_t)b * LL * DI;
    float* yb = ycon + ((size_t)k * BB + b) * LL * DI;
    for (int l = 0; l < LL; ++l) {
        int r = scan_row(k, l);
        const float* g = grow + (size_t)l * MROWS;
        __builtin_prefetch(g + 4 * MROWS, 0, 1);
        float u = xb[(size_t)r * DI + d];
        float xv = g[d] + bias;
        float dt = fmaxf(xv, 0.0f) + log1pf(__expf(-fabsf(xv)));   // softplus
        float du = dt * u;
        float y = 0.0f;
#pragma unroll
        for (int n = 0; n < DS; ++n) {
            float Bv = g[DI + n];
            float Cv = g[DI + DS + n];
            h[n] = __expf(dt * A[n]) * h[n] + du * Bv;
            y += h[n] * Cv;
        }
        yb[(size_t)r * DI + d] = y + u * Dp;
    }
}

// ---------------------------------------------------------------------------
// Sum 4 direction contributions, LayerNorm over 192 ch, gate with silu(z),
// emit bf16 rows for the final GEMM.
// ---------------------------------------------------------------------------
__global__ void k_ln_gate(const float* __restrict__ ycon, const float* __restrict__ z,
                          const float* __restrict__ lnw, const float* __restrict__ lnb,
                          __bf16* __restrict__ yg) {
    int row = blockIdx.x;             // 0..32767
    int d = threadIdx.x;              // 0..191
    __shared__ float red[DI];
    __shared__ float stats[2];
    float v = 0.0f;
#pragma unroll
    for (int k = 0; k < KD; ++k) v += ycon[((size_t)k * MTOT + row) * DI + d];
    red[d] = v;
    __syncthreads();
    if (d < 64) red[d] = red[d] + red[d + 64] + red[d + 128];
    __syncthreads();
    for (int s = 32; s > 0; s >>= 1) { if (d < s) red[d] += red[d + s]; __syncthreads(); }
    if (d == 0) stats[0] = red[0] * (1.0f / DI);
    __syncthreads();
    float mu = stats[0];
    float dv = v - mu;
    red[d] = dv * dv;
    __syncthreads();
    if (d < 64) red[d] = red[d] + red[d + 64] + red[d + 128];
    __syncthreads();
    for (int s = 32; s > 0; s >>= 1) { if (d < s) red[d] += red[d + s]; __syncthreads(); }
    if (d == 0) stats[1] = red[0] * (1.0f / DI);
    __syncthreads();
    float yn = dv * rsqrtf(stats[1] + 1e-5f) * lnw[d] + lnb[d];
    float zz = z[(size_t)row * DI + d];
    yg[(size_t)row * DI + d] = (__bf16)(yn * dev_silu(zz));
}

// ---------------------------------------------------------------------------
// GEMM 3: out = yg @ out_proj_w^T    M=32768, N=96, K=192  -> d_out f32
// ---------------------------------------------------------------------------
__global__ void k_outproj(const __bf16* __restrict__ yg, const __bf16* __restrict__ wobf,
                          float* __restrict__ out) {
    int lane = threadIdx.x & 31, wave = threadIdx.x >> 5;
    int mt = blockIdx.x * 4 + wave;   // 0..2047
    int nt = blockIdx.y;              // 0..5
    int r = lane & 15, half = lane >> 4;
    const __bf16* pa = yg + ((size_t)(mt * 16 + r)) * DI + 8 * half;
    const __bf16* pb = wobf + ((size_t)(nt * 16 + r)) * DI + 8 * half;
    v8f acc = wmma_bf16_tile<DI>(pa, pb);
    int nc = nt * 16 + r;
    int m0 = mt * 16 + half * 8;
#pragma unroll
    for (int i = 0; i < 8; ++i) out[(size_t)(m0 + i) * DM + nc] = acc[i];
}

// ---------------------------------------------------------------------------
extern "C" void kernel_launch(void* const* d_in, const int* in_sizes, int n_in,
                              void* d_out, int out_size, void* d_ws, size_t ws_size,
                              hipStream_t stream) {
    const float* x    = (const float*)d_in[0];
    const float* ipw  = (const float*)d_in[1];
    const float* cw   = (const float*)d_in[2];
    const float* cb   = (const float*)d_in[3];
    const float* xpw  = (const float*)d_in[4];
    const float* dtw  = (const float*)d_in[5];
    const float* dtb  = (const float*)d_in[6];
    const float* alog = (const float*)d_in[7];
    const float* Dsv  = (const float*)d_in[8];
    const float* lnw  = (const float*)d_in[9];
    const float* lnb  = (const float*)d_in[10];
    const float* opw  = (const float*)d_in[11];
    float* out = (float*)d_out;

    char* w = (char*)d_ws;
    size_t o = 0;
    auto take = [&](size_t bytes) -> char* {
        char* p = w + o;
        o = (o + bytes + 255) & ~(size_t)255;
        return p;
    };
    __bf16* xbf  = (__bf16*)take((size_t)MTOT * DM * 2);          // x in bf16
    __bf16* wbf  = (__bf16*)take((size_t)2 * DI * DM * 2);        // in_proj_w bf16
    __bf16* wobf = (__bf16*)take((size_t)DM * DI * 2);            // out_proj_w bf16
    __bf16* wall = (__bf16*)take((size_t)KD * MROWS * DI * 2);    // fused proj bf16
    float*  xq   = (float*)take((size_t)MTOT * DI * 4);           // pre-conv path
    float*  zbuf = (float*)take((size_t)MTOT * DI * 4);           // gate path
    float*  xcf  = (float*)take((size_t)MTOT * DI * 4);           // conv+silu f32
    __bf16* xcb  = (__bf16*)take((size_t)MTOT * DI * 2);          // conv+silu bf16
    float*  gout = (float*)take((size_t)BB * KD * LL * MROWS * 4);// delta|B|C per l
    float*  ycon = (float*)take((size_t)KD * MTOT * DI * 4);      // per-dir scan out
    __bf16* yg   = (__bf16*)take((size_t)MTOT * DI * 2);          // gated bf16

    // prep: bf16 conversions + fused projection matrix
    {
        int n = MTOT * DM;
        k_cvt_bf16<<<dim3((n + 255) / 256), dim3(256), 0, stream>>>(x, xbf, n);
    }
    {
        int n = 2 * DI * DM;
        k_cvt_bf16<<<dim3((n + 255) / 256), dim3(256), 0, stream>>>(ipw, wbf, n);
    }
    {
        int n = DM * DI;
        k_cvt_bf16<<<dim3((n + 255) / 256), dim3(256), 0, stream>>>(opw, wobf, n);
    }
    {
        int n = KD * MROWS * DI;
        k_build_wall<<<dim3((n + 255) / 256), dim3(256), 0, stream>>>(xpw, dtw, wall);
    }

    // 1) in_proj GEMM (WMMA)
    k_inproj<<<dim3(MTOT / 16, 24 / 4), dim3(128), 0, stream>>>(xbf, wbf, xq, zbuf);

    // 2) depthwise conv + SiLU
    {
        size_t n = (size_t)BB * LL * DI;
        k_conv<<<dim3((unsigned)((n + 255) / 256)), dim3(256), 0, stream>>>(xq, cw, cb, xcf, xcb);
    }

    // 3) fused x_proj/dt_proj GEMM per (b, direction) (WMMA)
    k_gemm2<<<dim3(BB * KD, MROWS / 16, (LL / 16) / 4), dim3(128), 0, stream>>>(xcb, wall, gout);

    // 4) selective scan (sequential recurrence, register-resident state)
    k_scan<<<dim3(BB * KD), dim3(DI), 0, stream>>>(gout, xcf, alog, Dsv, dtb, ycon);

    // 5) sum directions + LayerNorm + silu(z) gate
    k_ln_gate<<<dim3(MTOT), dim3(DI), 0, stream>>>(ycon, zbuf, lnw, lnb, yg);

    // 6) out_proj GEMM (WMMA) straight into d_out
    k_outproj<<<dim3(MTOT / 16 / 4, DM / 16), dim3(128), 0, stream>>>(yg, wobf, out);
}
